// HierarchicalSoftmax_90778428769076
// MI455X (gfx1250) — matrix operand order repaired and checked by reading, not compiled, not run on hardware
//
#include <hip/hip_runtime.h>

typedef __bf16 bf16_t;
typedef bf16_t v16bf __attribute__((ext_vector_type(16)));
typedef float  v8f   __attribute__((ext_vector_type(8)));
typedef float  v4f   __attribute__((ext_vector_type(4)));

#define BATCH   2048
#define OUTD    50257
#define LVL     225
#define NPAD    240                 // 15 tiles of 16
#define NT      15                  // col tiles per head
#define NCT     30                  // real col tiles (2 heads)
#define NCT_P   32                  // padded so every wave does 4 WMMAs (EXEC all-ones)
#define KTILES  1571                // ceil(50257 / 32)
#define KSPLIT  8
#define KT_PER  ((KTILES + KSPLIT - 1) / KSPLIT)   // 197
#define LROW    (2 * NPAD)          // 480 floats per batch row of logits/h

// ---------------------------------------------------------------------------
// Kernel 1: pack W0/W1 (f32, [50257][225]) into bf16 B-fragment tiles.
// Tile = 32(K) x 16(N) bf16 = 512 elements = 1 KB, laid out so that GEMM
// lane L reads its 16 fragment values as one contiguous 32B chunk:
//   element e: L = e>>4, j = e&15, K = (L<16 ? 0 : 8) + (j<8 ? j : j+8), N = L&15
// Tiles indexed [ct][kt], ct in [0,32): 0-14 = W0 n-tiles, 15-29 = W1,
// 30-31 = zero padding tiles.
// ---------------------------------------------------------------------------
__global__ __launch_bounds__(256) void pack_w(const float* __restrict__ W0,
                                              const float* __restrict__ W1,
                                              bf16_t* __restrict__ Wp) {
    const int tile = blockIdx.x;            // ct * KTILES + kt
    const int ct = tile / KTILES;
    const int kt = tile - ct * KTILES;
    const bool real = (ct < NCT);
    const float* W = (ct < NT) ? W0 : W1;
    const int nbase = (ct < NT ? ct : ct - NT) * 16;

    for (int e = threadIdx.x; e < 512; e += 256) {
        const int L = e >> 4;
        const int j = e & 15;
        const int K = ((L < 16) ? 0 : 8) + ((j < 8) ? j : j + 8);
        const int n = nbase + (L & 15);
        const int kk = kt * 32 + K;
        float v = 0.0f;
        if (real && kk < OUTD && n < LVL) v = W[(size_t)kk * LVL + n];
        Wp[(size_t)tile * 512 + e] = (bf16_t)v;
    }
}

// ---------------------------------------------------------------------------
// Kernel 2: dual GEMM (logits for both heads) with split-K.
// grid = (128 M-tiles, 8 K-slices), block = 256 (8 waves).
// Wave w handles column tiles {w, w+8, w+16, w+24} (padded to 32 tiles so the
// WMMA loop is branchless -> EXEC is all ones at every v_wmma).
// Partial sums stored (no atomics) to partials[ks][2048][480].
// ---------------------------------------------------------------------------
__global__ __launch_bounds__(256) void dual_gemm(const float* __restrict__ x,
                                                 const bf16_t* __restrict__ Wp,
                                                 float* __restrict__ partials) {
    const int lane = threadIdx.x & 31;
    const int wid  = threadIdx.x >> 5;
    const int mt = blockIdx.x;
    const int ks = blockIdx.y;

    const int kt0 = ks * KT_PER;
    const int kt1 = (kt0 + KT_PER < KTILES) ? (kt0 + KT_PER) : KTILES;

    // A-fragment addressing (16x32 bf16 A layout):
    //   lane holds row M = lane&15; lane<16 -> K {0..7,16..23}, lane>=16 -> +8
    const int rowA = (mt << 4) + (lane & 15);
    const int kbA  = (lane < 16) ? 0 : 8;
    const float* xrow = x + (size_t)rowA * OUTD;

    int cols[4];
    v8f acc[4] = {};
#pragma unroll
    for (int j = 0; j < 4; ++j) cols[j] = wid + 8 * j;   // < 32 always

    for (int kt = kt0; kt < kt1; ++kt) {
        const int k0 = kt * 32;

        // prefetch the x stream a few tiles ahead
        __builtin_prefetch(xrow + k0 + 256, 0, 1);

        // ---- load + convert A fragment (f32 -> bf16) ----
        float fa[16];
        if (k0 + 32 <= OUTD) {
            const float* p0 = xrow + k0 + kbA;
            *(v4f*)(fa + 0)  = *(const v4f*)(p0 + 0);    // K kbA+0..3
            *(v4f*)(fa + 4)  = *(const v4f*)(p0 + 4);    // K kbA+4..7
            *(v4f*)(fa + 8)  = *(const v4f*)(p0 + 16);   // K kbA+16..19
            *(v4f*)(fa + 12) = *(const v4f*)(p0 + 20);   // K kbA+20..23
        } else {
#pragma unroll
            for (int h = 0; h < 16; ++h) {
                const int kk = k0 + kbA + (h < 8 ? h : h + 8);
                fa[h] = (kk < OUTD) ? xrow[kk] : 0.0f;
            }
        }
        v16bf a;
#pragma unroll
        for (int h = 0; h < 16; ++h) a[h] = (bf16_t)fa[h];

        // ---- 4 column tiles: load pre-swizzled B, WMMA ----
#pragma unroll
        for (int j = 0; j < 4; ++j) {
            const v16bf b = *(const v16bf*)(Wp +
                (((size_t)cols[j] * KTILES + kt) << 9) + ((size_t)lane << 4));
            acc[j] = __builtin_amdgcn_wmma_f32_16x16x32_bf16(
                false, a, false, b, (short)0, acc[j], false, false);
        }
    }

    // ---- store partial C tiles (32-bit C layout: lane<16 -> M v+0..7, N=lane;
    //      lane>=16 -> M v+8, N=lane-16) ----
    float* prow = partials + (size_t)ks * BATCH * LROW;
#pragma unroll
    for (int j = 0; j < 4; ++j) {
        const int ct = cols[j];
        if (ct >= NCT) continue;                 // padding tiles: drop
        const int head = (ct >= NT) ? 1 : 0;
        const int ncol = (ct - head * NT) * 16 + (lane & 15);
        const int mrow = (mt << 4) + ((lane < 16) ? 0 : 8);
#pragma unroll
        for (int v = 0; v < 8; ++v) {
            prow[(size_t)(mrow + v) * LROW + head * NPAD + ncol] = acc[j][v];
        }
    }
}

// ---------------------------------------------------------------------------
// Kernel 3: reduce split-K partials (fixed order -> deterministic), add bias,
// masked softmax over 225 valid cols. One wave per (batch row, head).
// ---------------------------------------------------------------------------
__global__ __launch_bounds__(64) void softmax_rows(const float* __restrict__ partials,
                                                   const float* __restrict__ b0,
                                                   const float* __restrict__ b1,
                                                   float* __restrict__ hbuf) {
    const int b = blockIdx.x;
    const int head = threadIdx.x >> 5;
    const int lane = threadIdx.x & 31;
    const float* bias = head ? b1 : b0;
    float* hrow = hbuf + (size_t)b * LROW + head * NPAD;
    const size_t base = (size_t)b * LROW + head * NPAD;

    float v[8];
    float mx = -3.402823466e38f;
#pragma unroll
    for (int i = 0; i < 8; ++i) {
        const int n = i * 32 + lane;
        float t = -3.402823466e38f;
        if (n < LVL) {
            float s = 0.0f;
#pragma unroll
            for (int ks = 0; ks < KSPLIT; ++ks)
                s += partials[(size_t)ks * BATCH * LROW + base + n];
            t = s + bias[n];
        }
        v[i] = t;
        mx = fmaxf(mx, t);
    }
#pragma unroll
    for (int off = 16; off; off >>= 1) mx = fmaxf(mx, __shfl_xor(mx, off, 32));

    float sum = 0.0f;
#pragma unroll
    for (int i = 0; i < 8; ++i) {
        const int n = i * 32 + lane;
        const float e = (n < LVL) ? __expf(v[i] - mx) : 0.0f;
        v[i] = e;
        sum += e;
    }
#pragma unroll
    for (int off = 16; off; off >>= 1) sum += __shfl_xor(sum, off, 32);

    const float inv = 1.0f / sum;
#pragma unroll
    for (int i = 0; i < 8; ++i) {
        const int n = i * 32 + lane;
        if (n < NPAD) hrow[n] = v[i] * inv;
    }
}

// ---------------------------------------------------------------------------
// Kernel 4: per-row outer product, truncated to OUTD columns.
// out[b, i] = h0[b, i/225] * h1[b, i%225]. h rows staged in LDS.
// ---------------------------------------------------------------------------
__global__ __launch_bounds__(256) void outer_product(const float* __restrict__ hbuf,
                                                     float* __restrict__ out) {
    const int b = blockIdx.x;
    __shared__ float hs[LROW];
    for (int i = threadIdx.x; i < LROW; i += 256)
        hs[i] = hbuf[(size_t)b * LROW + i];
    __syncthreads();

    float* orow = out + (size_t)b * OUTD;
    for (int i = threadIdx.x; i < OUTD; i += 256) {
        const unsigned q = (unsigned)i / 225u;
        const unsigned r = (unsigned)i - q * 225u;
        orow[i] = hs[q] * hs[NPAD + r];
    }
}

// ---------------------------------------------------------------------------
// Workspace layout (bytes):
//   [0, 3.93MB)            h      : 2048 * 480 f32
//   [3.93MB, 35.39MB)      parts  : 8 * 2048 * 480 f32
//   [35.39MB, 86.87MB)     Wp     : 32 * 1571 tiles * 1KB bf16
// ---------------------------------------------------------------------------
extern "C" void kernel_launch(void* const* d_in, const int* in_sizes, int n_in,
                              void* d_out, int out_size, void* d_ws, size_t ws_size,
                              hipStream_t stream) {
    const float* x  = (const float*)d_in[0];
    const float* W0 = (const float*)d_in[1];
    const float* W1 = (const float*)d_in[2];
    const float* b0 = (const float*)d_in[3];
    const float* b1 = (const float*)d_in[4];
    float* out = (float*)d_out;

    const size_t H_BYTES = (size_t)BATCH * LROW * sizeof(float);          // 3,932,160
    float*  hbuf     = (float*)d_ws;
    float*  partials = (float*)((char*)d_ws + H_BYTES);
    bf16_t* Wp       = (bf16_t*)((char*)d_ws + H_BYTES * (1 + KSPLIT));

    pack_w<<<NCT_P * KTILES, 256, 0, stream>>>(W0, W1, Wp);
    dual_gemm<<<dim3(BATCH / 16, KSPLIT), 256, 0, stream>>>(x, Wp, partials);
    softmax_rows<<<BATCH, 64, 0, stream>>>(partials, b0, b1, hbuf);
    outer_product<<<BATCH, 256, 0, stream>>>(hbuf, out);
}
